// SSM_41377714929826
// MI455X (gfx1250) — matrix operand order repaired and checked by reading, not compiled, not run on hardware
//
#include <hip/hip_runtime.h>

// ---------------------------------------------------------------------------
// Selective SSM (Mamba-like) for MI455X / gfx1250, wave32 + WMMA.
//   B=8, S=4096, D=64, ds=16.
// Pipeline:
//   k1 repack   : W_A/W_B/W_C/b_B/b_C -> bf16 B-operand fragment layout
//   k2 gemmABx  : A = x@W_A (bf16 WMMA) ; Bx = (x(x)x)@WB2 (bf16 WMMA)
//   k3a chunk   : per 64-step chunk, P_c / s_c via v_wmma_f32_16x16x4_f32,
//                 A_t double-buffered in LDS via global_load_async_to_lds_b128
//   k3b combine : sequential over 64 chunks/batch -> chunk start states
//   k3c inner   : replay chunk-local scan, emit all h_t
//   k4 output   : out = (x(x)h)@WC2 + h@BC2 (bf16 WMMA)
// ---------------------------------------------------------------------------

typedef __attribute__((ext_vector_type(16))) __bf16 v16bf;
typedef __attribute__((ext_vector_type(8)))  float  v8f;
typedef __attribute__((ext_vector_type(2)))  float  v2f;

static constexpr int Bb  = 8;
static constexpr int Ss  = 4096;
static constexpr int TOK = Bb * Ss;        // 32768 tokens
static constexpr int CH  = 64;             // scan chunk length
static constexpr int NCH = Ss / CH;        // 64 chunks per batch
static constexpr int NCHT = TOK / CH;      // 512 chunks total

// workspace byte offsets (total ~36.9 MiB)
static constexpr size_t OFF_A   = 0;                                  // TOK*256 f32
static constexpr size_t OFF_BX  = OFF_A   + (size_t)TOK * 256 * 4;    // TOK*16 f32
static constexpr size_t OFF_H   = OFF_BX  + (size_t)TOK * 16 * 4;     // TOK*16 f32
static constexpr size_t OFF_P   = OFF_H   + (size_t)TOK * 16 * 4;     // NCHT*256 f32
static constexpr size_t OFF_S   = OFF_P   + (size_t)NCHT * 256 * 4;   // NCHT*16 f32
static constexpr size_t OFF_HS  = OFF_S   + (size_t)NCHT * 16 * 4;    // NCHT*16 f32
static constexpr size_t OFF_WAB = OFF_HS  + (size_t)NCHT * 16 * 4;    // 16384 bf16
static constexpr size_t OFF_WB2 = OFF_WAB + (size_t)16384 * 2;        // 65536 bf16
static constexpr size_t OFF_WC2 = OFF_WB2 + (size_t)65536 * 2;        // 65536 bf16
static constexpr size_t OFF_BB2 = OFF_WC2 + (size_t)65536 * 2;        // 1024  bf16
static constexpr size_t OFF_BC2 = OFF_BB2 + (size_t)1024  * 2;        // 2048  bf16

__device__ __forceinline__ v8f wmma_bf16(v16bf a, v16bf b, v8f c) {
    return __builtin_amdgcn_wmma_f32_16x16x32_bf16(false, a, false, b,
                                                   (short)0, c, false, false);
}
__device__ __forceinline__ v8f wmma_f32(v2f a, v2f b, v8f c) {
    return __builtin_amdgcn_wmma_f32_16x16x4_f32(false, a, false, b,
                                                 (short)0, c, false, false);
}

// place a float4 into 4 consecutive bf16 vector elements (constant E)
#define PUT4(V, E, F4)                                                       \
    {                                                                        \
        (V)[(E) + 0] = (__bf16)(F4).x;                                       \
        (V)[(E) + 1] = (__bf16)(F4).y;                                       \
        (V)[(E) + 2] = (__bf16)(F4).z;                                       \
        (V)[(E) + 3] = (__bf16)(F4).w;                                       \
    }

// Async copy of 1KB (32 lanes x 32B) from global to LDS; tracked by ASYNCcnt.
__device__ __forceinline__ void async_copy_1kb(uint32_t lds_base,
                                               const float* gbase, int lane) {
    uint64_t ga = (uint64_t)(uintptr_t)gbase + (uint64_t)lane * 32;
    uint32_t la = lds_base + (uint32_t)lane * 32;
    asm volatile("global_load_async_to_lds_b128 %0, %1, off"
                 :: "v"(la), "v"(ga) : "memory");
    asm volatile("global_load_async_to_lds_b128 %0, %1, off offset:16"
                 :: "v"(la), "v"(ga) : "memory");
}

// ---------------------------------------------------------------------------
// k1: repack weights into bf16 WMMA B-operand fragments.
// B-fragment layout (32x16, bf16): element e, lane l -> K=(l&16?8:0)+(e>>3)*16+(e&7), N=l&15
// ---------------------------------------------------------------------------
__global__ void repack_kernel(const float* __restrict__ WA,
                              const float* __restrict__ WB,
                              const float* __restrict__ WC,
                              const float* __restrict__ bB,
                              const float* __restrict__ bC,
                              __bf16* __restrict__ wab, __bf16* __restrict__ wb2,
                              __bf16* __restrict__ wc2, __bf16* __restrict__ bb2,
                              __bf16* __restrict__ bc2) {
    const int T0 = 16384, T1 = 65536, T2 = 65536, T3 = 1024, T4 = 2048;
    const int total = T0 + T1 + T2 + T3 + T4;
    int stride = gridDim.x * blockDim.x;
    for (int id = blockIdx.x * blockDim.x + threadIdx.x; id < total; id += stride) {
        int r = id;
        if (r < T0) {  // W_A [64 x 256]: 2 K-tiles x 16 N-tiles
            int e = r & 15, lane = (r >> 4) & 31, fn = r >> 9;
            int kap = fn >> 4, nu = fn & 15, hi = (lane >> 4) & 1;
            int Kg = kap * 32 + hi * 8 + (e >> 3) * 16 + (e & 7);
            int Ng = nu * 16 + (lane & 15);
            wab[id] = (__bf16)WA[Kg * 256 + Ng];
            continue;
        }
        r -= T0;
        if (r < T1) {  // WB2[j=k*64+d, n] = W_B[k, n*64+d] : 128 K-tiles
            int e = r & 15, lane = (r >> 4) & 31, kt = r >> 9;
            int hi = (lane >> 4) & 1;
            int Kg = kt * 32 + hi * 8 + (e >> 3) * 16 + (e & 7);      // j in [0,4096)
            int k = Kg >> 6, d = Kg & 63, n = lane & 15;
            wb2[r] = (__bf16)WB[k * 1024 + n * 64 + d];
            continue;
        }
        r -= T1;
        if (r < T2) {  // WC2[j=k*16+n, d] = W_C[k, n*64+d] : 32 K-tiles x 4 N-tiles
            int e = r & 15, lane = (r >> 4) & 31, fn = r >> 9;
            int kt = fn >> 2, nu = fn & 3, hi = (lane >> 4) & 1;
            int Kg = kt * 32 + hi * 8 + (e >> 3) * 16 + (e & 7);      // j in [0,1024)
            int k = Kg >> 4, n = Kg & 15;
            int d = nu * 16 + (lane & 15);
            wc2[r] = (__bf16)WC[k * 1024 + n * 64 + d];
            continue;
        }
        r -= T2;
        if (r < T3) {  // BB2[d, n] = b_B[n*64+d] : 2 K-tiles
            int e = r & 15, lane = (r >> 4) & 31, kap = r >> 9;
            int hi = (lane >> 4) & 1;
            int Kd = kap * 32 + hi * 8 + (e >> 3) * 16 + (e & 7);     // d in [0,64)
            int n = lane & 15;
            bb2[r] = (__bf16)bB[n * 64 + Kd];
            continue;
        }
        r -= T3;
        {  // BC2[n, d] = b_C[n*64+d] : 1 K-tile (rows >=16 zero) x 4 N-tiles
            int e = r & 15, lane = (r >> 4) & 31, nu = r >> 9;
            int hi = (lane >> 4) & 1;
            int Kr = hi * 8 + (e >> 3) * 16 + (e & 7);
            int d = nu * 16 + (lane & 15);
            bc2[r] = (Kr < 16) ? (__bf16)bC[Kr * 64 + d] : (__bf16)0.0f;
        }
    }
}

// ---------------------------------------------------------------------------
// k2: per 16-token tile (one wave):  A = x@W_A + b_A ;  Bx = XX@WB2 + x@BB2
// A-fragment (16x32, bf16): element e, lane l -> M=l&15, K=(l&16?8:0)+(e>>3)*16+(e&7)
// ---------------------------------------------------------------------------
__global__ __launch_bounds__(256) void gemm_abx_kernel(
        const float* __restrict__ x, const float* __restrict__ bA,
        const __bf16* __restrict__ wab, const __bf16* __restrict__ wb2,
        const __bf16* __restrict__ bb2,
        float* __restrict__ Aout, float* __restrict__ Bxout) {
    int lane = threadIdx.x & 31;
    int wave = threadIdx.x >> 5;
    int tile = blockIdx.x * 8 + wave;          // 0..2047
    int t0 = tile * 16;
    int half = (lane >> 4) & 1, col = lane & 15, mbase = half * 8;
    int t = t0 + col;                           // this lane's A-operand row token
    const float* xrow = x + (size_t)t * 64;

    // ---- A-operand fragments of x: 4 runs of 8 floats at mbase + {0,16,32,48}
    const float* xf = xrow + mbase;
    float4 r0a = *(const float4*)(xf + 0),  r0b = *(const float4*)(xf + 4);
    float4 r1a = *(const float4*)(xf + 16), r1b = *(const float4*)(xf + 20);
    float4 r2a = *(const float4*)(xf + 32), r2b = *(const float4*)(xf + 36);
    float4 r3a = *(const float4*)(xf + 48), r3b = *(const float4*)(xf + 52);
    v16bf fx0, fx1;
    PUT4(fx0, 0, r0a) PUT4(fx0, 4, r0b) PUT4(fx0, 8, r1a) PUT4(fx0, 12, r1b)
    PUT4(fx1, 0, r2a) PUT4(fx1, 4, r2b) PUT4(fx1, 8, r3a) PUT4(fx1, 12, r3b)

    // ---- packed bf16 copy of the full row (static indices only)
    const float4* xp = (const float4*)xrow;
    v16bf xk0, xk1, xk2, xk3;
    {
        float4 q;
        q = xp[0];  PUT4(xk0, 0, q)  q = xp[1];  PUT4(xk0, 4, q)
        q = xp[2];  PUT4(xk0, 8, q)  q = xp[3];  PUT4(xk0, 12, q)
        q = xp[4];  PUT4(xk1, 0, q)  q = xp[5];  PUT4(xk1, 4, q)
        q = xp[6];  PUT4(xk1, 8, q)  q = xp[7];  PUT4(xk1, 12, q)
        q = xp[8];  PUT4(xk2, 0, q)  q = xp[9];  PUT4(xk2, 4, q)
        q = xp[10]; PUT4(xk2, 8, q)  q = xp[11]; PUT4(xk2, 12, q)
        q = xp[12]; PUT4(xk3, 0, q)  q = xp[13]; PUT4(xk3, 4, q)
        q = xp[14]; PUT4(xk3, 8, q)  q = xp[15]; PUT4(xk3, 12, q)
    }

    // ---- A = x @ W_A + b_A  (16 N-tiles x 2 K-tiles) ----
    const v16bf* wa = (const v16bf*)wab;
#pragma unroll
    for (int nu = 0; nu < 16; nu++) {
        v8f acc = {};
        acc = wmma_bf16(fx0, wa[(0 * 16 + nu) * 32 + lane], acc);
        acc = wmma_bf16(fx1, wa[(1 * 16 + nu) * 32 + lane], acc);
        float bias = bA[nu * 16 + col];
#pragma unroll
        for (int v = 0; v < 8; v++) {
            int m = mbase + v;
            Aout[(size_t)(t0 + m) * 256 + nu * 16 + col] = acc[v] + bias;
        }
    }

    // ---- Bx = XX @ WB2 + x @ BB2  (128 + 2 K-tiles, N=16) ----
    const v16bf* wb = (const v16bf*)wb2;
    v8f acc = {};
#pragma unroll
    for (int g = 0; g < 4; g++) {
        v16bf xkv = (g == 0) ? xk0 : (g == 1) ? xk1 : (g == 2) ? xk2 : xk3;
#pragma unroll
        for (int j = 0; j < 32; j++) {            // kt = g*32 + j
            int kt = g * 32 + j;
            __bf16 xk = xkv[j >> 1];              // x[t, kt>>1], constant extract
            v16bf base = (j & 1) ? fx1 : fx0;
            v16bf frag = base * xk;               // v_pk_mul_bf16 x8
            acc = wmma_bf16(frag, wb[kt * 32 + lane], acc);
        }
    }
    const v16bf* bb = (const v16bf*)bb2;
    acc = wmma_bf16(fx0, bb[0 * 32 + lane], acc);
    acc = wmma_bf16(fx1, bb[1 * 32 + lane], acc);
#pragma unroll
    for (int v = 0; v < 8; v++)
        Bxout[(size_t)(t0 + mbase + v) * 16 + col] = acc[v];
}

// ---------------------------------------------------------------------------
// k3a: per chunk c (512 blocks x 1 wave): P_c = A_63...A_0 ; s_c (col 0 of U)
// via f32 WMMA 16x16x4. A_t is double-buffered in LDS with async loads so the
// global-load latency is off the 64-step serial chain.
// ---------------------------------------------------------------------------
__global__ __launch_bounds__(32) void chunk_scan_kernel(
        const float* __restrict__ Aw, const float* __restrict__ Bx,
        float* __restrict__ Pout, float* __restrict__ sout) {
    int c = blockIdx.x;
    int lane = threadIdx.x & 31;
    int half = (lane >> 4) & 1, col = lane & 15, mbase = half * 8;
    int am = lane & 15;          // A-operand row index
    int t0 = c * CH;

    __shared__ float Pl[256];
    __shared__ float Ul[256];
    __shared__ float Abuf[2][256];

    uint32_t abase[2];
    abase[0] = (uint32_t)(uintptr_t)&Abuf[0][0];
    abase[1] = (uint32_t)(uintptr_t)&Abuf[1][0];

    // prologue: async-fetch A for step 0 into buffer 0
    async_copy_1kb(abase[0], Aw + (size_t)t0 * 256, lane);

    v8f P, U;
#pragma unroll
    for (int v = 0; v < 8; v++) {
        P[v] = ((mbase + v) == col) ? 1.0f : 0.0f;   // identity
        U[v] = 0.0f;
    }

    for (int s = 0; s < CH; s++) {
        int t = t0 + s;
        // prefetch A for step s+1 (clamped on last step) into the other buffer
        int tn = t0 + ((s + 1 < CH) ? (s + 1) : s);
        async_copy_1kb(abase[(s + 1) & 1], Aw + (size_t)tn * 256, lane);

#pragma unroll
        for (int v = 0; v < 8; v++) {
            Pl[(mbase + v) * 16 + col] = P[v];
            Ul[(mbase + v) * 16 + col] = U[v];
        }
        __syncthreads();

        // wait until the two async loads for THIS step's buffer have landed
        // (async loads complete in order; <=2 outstanding leaves only s+1's)
        asm volatile("s_wait_asynccnt 2" ::: "memory");

        const float* Ards = &Abuf[s & 1][0];
        v8f pn = {}, un = {};
#pragma unroll
        for (int q = 0; q < 4; q++) {
            int koff = q * 4 + half * 2;
            v2f a;
            a[0] = Ards[am * 16 + koff + 0];
            a[1] = Ards[am * 16 + koff + 1];
            v2f bp, bu;
#pragma unroll
            for (int v = 0; v < 2; v++) {
                int Kr = q * 4 + half * 2 + v;
                bp[v] = Pl[Kr * 16 + col];
                bu[v] = Ul[Kr * 16 + col];
            }
            pn = wmma_f32(a, bp, pn);
            un = wmma_f32(a, bu, un);
        }
        __syncthreads();
#pragma unroll
        for (int v = 0; v < 8; v++) {
            float bx = Bx[(size_t)t * 16 + mbase + v];
            un[v] += (col == 0) ? bx : 0.0f;   // s lives in column 0 of U
        }
        P = pn;
        U = un;
    }
    // drain remaining async traffic before kernel end
    asm volatile("s_wait_asynccnt 0" ::: "memory");

#pragma unroll
    for (int v = 0; v < 8; v++)
        Pout[(size_t)c * 256 + (mbase + v) * 16 + col] = P[v];
    if (col == 0) {
#pragma unroll
        for (int v = 0; v < 8; v++)
            sout[(size_t)c * 16 + mbase + v] = U[v];
    }
}

// ---------------------------------------------------------------------------
// k3b: per batch (8 blocks): sequential over 64 chunks: h <- P_c h + s_c
// ---------------------------------------------------------------------------
__global__ __launch_bounds__(32) void combine_kernel(
        const float* __restrict__ P, const float* __restrict__ s,
        float* __restrict__ hstart) {
    int b = blockIdx.x;
    int i = threadIdx.x;
    __shared__ float hl[16];
    float h = 0.0f;
    for (int c = 0; c < NCH; c++) {
        int g = b * NCH + c;
        if (i < 16 && c + 1 < NCH)
            __builtin_prefetch(P + (size_t)(g + 1) * 256 + i * 16, 0, 0);
        if (i < 16) {
            hstart[(size_t)g * 16 + i] = h;   // state entering chunk g
            hl[i] = h;
        }
        __syncthreads();
        if (i < 16) {
            const float* Pr = P + (size_t)g * 256 + i * 16;
            float acc = s[(size_t)g * 16 + i];
#pragma unroll
            for (int j = 0; j < 16; j++) acc += Pr[j] * hl[j];
            h = acc;
        }
        __syncthreads();
    }
}

// ---------------------------------------------------------------------------
// k3c: per chunk (512 blocks): replay local scan, emit h_t (post-update)
// ---------------------------------------------------------------------------
__global__ __launch_bounds__(32) void inner_scan_kernel(
        const float* __restrict__ Aw, const float* __restrict__ Bx,
        const float* __restrict__ hstart, float* __restrict__ hout) {
    int c = blockIdx.x;
    int i = threadIdx.x;
    __shared__ float hl[16];
    float h = (i < 16) ? hstart[(size_t)c * 16 + i] : 0.0f;
    int t0 = c * CH;
    for (int s = 0; s < CH; s++) {
        int t = t0 + s;
        if (i < 16 && s + 1 < CH)
            __builtin_prefetch(Aw + (size_t)(t + 1) * 256 + i * 16, 0, 0);
        if (i < 16) hl[i] = h;
        __syncthreads();
        if (i < 16) {
            const float* Ar = Aw + (size_t)t * 256 + i * 16;
            float acc = Bx[(size_t)t * 16 + i];
#pragma unroll
            for (int j = 0; j < 16; j++) acc += Ar[j] * hl[j];
            h = acc;
            hout[(size_t)t * 16 + i] = h;
        }
        __syncthreads();
    }
}

// ---------------------------------------------------------------------------
// k4: out = (x (x) h) @ WC2 + h @ BC2   (32 K-tiles x 4 N-tiles + bias tile)
// ---------------------------------------------------------------------------
__global__ __launch_bounds__(256) void output_kernel(
        const float* __restrict__ x, const float* __restrict__ h,
        const __bf16* __restrict__ wc2, const __bf16* __restrict__ bc2,
        float* __restrict__ out) {
    int lane = threadIdx.x & 31;
    int wave = threadIdx.x >> 5;
    int tile = blockIdx.x * 8 + wave;
    int t0 = tile * 16;
    int half = (lane >> 4) & 1, col = lane & 15, mbase = half * 8;
    int t = t0 + col;
    const float* xrow = x + (size_t)t * 64;

    // h fragment: h[t, mbase..mbase+7], lane-dependent offset in the ADDRESS
    const float* hrow = h + (size_t)t * 16 + mbase;
    float4 h0 = *(const float4*)(hrow + 0);
    float4 h1 = *(const float4*)(hrow + 4);
    v16bf hb, hz;
    PUT4(hb, 0, h0) PUT4(hb, 4, h1)
#pragma unroll
    for (int e = 8; e < 16; e++) hb[e] = hb[e - 8];   // n repeats for K>=16
#pragma unroll
    for (int e = 0; e < 16; e++) hz[e] = (e < 8) ? hb[e] : (__bf16)0.0f;

    // packed bf16 full x row (static indices only)
    const float4* xp = (const float4*)xrow;
    v16bf xk0, xk1, xk2, xk3;
    {
        float4 q;
        q = xp[0];  PUT4(xk0, 0, q)  q = xp[1];  PUT4(xk0, 4, q)
        q = xp[2];  PUT4(xk0, 8, q)  q = xp[3];  PUT4(xk0, 12, q)
        q = xp[4];  PUT4(xk1, 0, q)  q = xp[5];  PUT4(xk1, 4, q)
        q = xp[6];  PUT4(xk1, 8, q)  q = xp[7];  PUT4(xk1, 12, q)
        q = xp[8];  PUT4(xk2, 0, q)  q = xp[9];  PUT4(xk2, 4, q)
        q = xp[10]; PUT4(xk2, 8, q)  q = xp[11]; PUT4(xk2, 12, q)
        q = xp[12]; PUT4(xk3, 0, q)  q = xp[13]; PUT4(xk3, 4, q)
        q = xp[14]; PUT4(xk3, 8, q)  q = xp[15]; PUT4(xk3, 12, q)
    }

    v8f acc0 = {}, acc1 = {}, acc2 = {}, acc3 = {};
    const v16bf* wc = (const v16bf*)wc2;
#pragma unroll
    for (int g = 0; g < 4; g++) {                 // kt = g*8 + j
        v16bf xkv = (g == 0) ? xk0 : (g == 1) ? xk1 : (g == 2) ? xk2 : xk3;
#pragma unroll
        for (int j = 0; j < 8; j++) {
            int kt = g * 8 + j;
            __bf16 xa = xkv[2 * j + 0];           // x[t, 2kt]
            __bf16 xb = xkv[2 * j + 1];           // x[t, 2kt+1]
            v16bf xs = {xa, xa, xa, xa, xa, xa, xa, xa,
                        xb, xb, xb, xb, xb, xb, xb, xb};
            v16bf frag = xs * hb;                 // G[t, j] = x_k * h_n
            acc0 = wmma_bf16(frag, wc[(kt * 4 + 0) * 32 + lane], acc0);
            acc1 = wmma_bf16(frag, wc[(kt * 4 + 1) * 32 + lane], acc1);
            acc2 = wmma_bf16(frag, wc[(kt * 4 + 2) * 32 + lane], acc2);
            acc3 = wmma_bf16(frag, wc[(kt * 4 + 3) * 32 + lane], acc3);
        }
    }
    const v16bf* bc = (const v16bf*)bc2;
    acc0 = wmma_bf16(hz, bc[0 * 32 + lane], acc0);
    acc1 = wmma_bf16(hz, bc[1 * 32 + lane], acc1);
    acc2 = wmma_bf16(hz, bc[2 * 32 + lane], acc2);
    acc3 = wmma_bf16(hz, bc[3 * 32 + lane], acc3);

#pragma unroll
    for (int v = 0; v < 8; v++) {
        size_t row = (size_t)(t0 + mbase + v) * 64;
        out[row +  0 + col] = acc0[v];
        out[row + 16 + col] = acc1[v];
        out[row + 32 + col] = acc2[v];
        out[row + 48 + col] = acc3[v];
    }
}

// ---------------------------------------------------------------------------
extern "C" void kernel_launch(void* const* d_in, const int* in_sizes, int n_in,
                              void* d_out, int out_size, void* d_ws, size_t ws_size,
                              hipStream_t stream) {
    const float* x  = (const float*)d_in[0];
    const float* WA = (const float*)d_in[1];
    const float* bA = (const float*)d_in[2];
    const float* WB = (const float*)d_in[3];
    const float* bB = (const float*)d_in[4];
    const float* WC = (const float*)d_in[5];
    const float* bC = (const float*)d_in[6];
    float* out = (float*)d_out;

    char* ws = (char*)d_ws;
    float*  Aout  = (float*)(ws + OFF_A);
    float*  Bx    = (float*)(ws + OFF_BX);
    float*  Hh    = (float*)(ws + OFF_H);
    float*  Pc    = (float*)(ws + OFF_P);
    float*  Sc    = (float*)(ws + OFF_S);
    float*  Hs    = (float*)(ws + OFF_HS);
    __bf16* wab   = (__bf16*)(ws + OFF_WAB);
    __bf16* wb2   = (__bf16*)(ws + OFF_WB2);
    __bf16* wc2   = (__bf16*)(ws + OFF_WC2);
    __bf16* bb2   = (__bf16*)(ws + OFF_BB2);
    __bf16* bc2   = (__bf16*)(ws + OFF_BC2);

    repack_kernel<<<588, 256, 0, stream>>>(WA, WB, WC, bB, bC,
                                           wab, wb2, wc2, bb2, bc2);
    gemm_abx_kernel<<<TOK / 128, 256, 0, stream>>>(x, bA, wab, wb2, bb2,
                                                   Aout, Bx);
    chunk_scan_kernel<<<NCHT, 32, 0, stream>>>(Aout, Bx, Pc, Sc);
    combine_kernel<<<Bb, 32, 0, stream>>>(Pc, Sc, Hs);
    inner_scan_kernel<<<NCHT, 32, 0, stream>>>(Aout, Bx, Hs, Hh);
    output_kernel<<<TOK / 128, 256, 0, stream>>>(x, Hh, wc2, bc2, out);
}
